// OptimizedWordGCN_57604101374325
// MI455X (gfx1250) — matrix operand order repaired and checked by reading, not compiled, not run on hardware
//
#include <hip/hip_runtime.h>
#include <hip/hip_bf16.h>

// ---------------------------------------------------------------------------
// TextGCN forward on MI455X (gfx1250, wave32).
// Dense GEMMs -> v_wmma_f32_16x16x32_bf16 (bf16 operands in memory, f32 acc).
// SpMM -> COO gather + global_atomic_add_f32, then one flat f32->bf16 pass.
// Row stride KP=320 everywhere (K padded 300->320 => 10 k-steps of 32).
// Each GEMM wave computes a 16x32 strip: A fragment reused across 2 WMMAs.
// ---------------------------------------------------------------------------

#define NW   30000
#define ND   8192
#define DIM  300
#define D2   150
#define KP   320        // padded K / row stride
#define NP2  160        // padded D2
#define NNZA 600000
#define NNZX 524288
#define ALPHA   0.7f
#define BN_SCALE 0.99999500003749969f   // 1/sqrt(1+1e-5)

typedef __attribute__((ext_vector_type(16))) __bf16 v16bf;
typedef __attribute__((ext_vector_type(8)))  __bf16 v8bf;   // 16 bytes
typedef __attribute__((ext_vector_type(8)))  float  v8f;

union BFrag { v16bf v; v8bf h[2]; };

static __device__ __forceinline__ unsigned short f32_to_bf16_bits(float f) {
    union { float f; unsigned int u; } x; x.f = f;
    unsigned int b = x.u;
    b += 0x7FFFu + ((b >> 16) & 1u);        // round-to-nearest-even
    return (unsigned short)(b >> 16);
}
static __device__ __forceinline__ float bf16_bits_to_f32(unsigned short u) {
    union { unsigned int i; float f; } x; x.i = ((unsigned int)u) << 16;
    return x.f;
}
static __device__ __forceinline__ unsigned int pack2(float a, float b) {
    return (unsigned int)f32_to_bf16_bits(a) |
           ((unsigned int)f32_to_bf16_bits(b) << 16);
}

// --------------------------- utility kernels -------------------------------

__global__ void zero_f32_kernel(float* p, long long n) {
    long long t = (long long)blockIdx.x * blockDim.x + threadIdx.x;
    if (t < n) p[t] = 0.0f;
}

// dst(bf16)[Rp x Cp] = src(f32)[R x C], zero padded
__global__ void pad_cvt_bf16_kernel(const float* __restrict__ src, int R, int C,
                                    unsigned short* __restrict__ dst,
                                    int Rp, int Cp) {
    long long t = (long long)blockIdx.x * blockDim.x + threadIdx.x;
    long long tot = (long long)Rp * Cp;
    if (t >= tot) return;
    int r = (int)(t / Cp);
    int c = (int)(t % Cp);
    float v = (r < R && c < C) ? src[(long long)r * C + c] : 0.0f;
    dst[t] = f32_to_bf16_bits(v);
}

// flat f32 -> bf16, 8 elements per thread (n multiple of 8)
__global__ void cvt_f32_bf16_kernel(const float* __restrict__ src,
                                    unsigned short* __restrict__ dst,
                                    long long n) {
    long long t = ((long long)blockIdx.x * blockDim.x + threadIdx.x) * 8;
    if (t >= n) return;
    const float4* s = (const float4*)(src + t);
    float4 a = s[0], b = s[1];
    uint4 o;
    o.x = pack2(a.x, a.y); o.y = pack2(a.z, a.w);
    o.z = pack2(b.x, b.y); o.w = pack2(b.z, b.w);
    *(uint4*)(dst + t) = o;
}

// fold eval-BatchNorm + linear bias into per-column scale/bias:
//   y = (dot + lin_b) * (g*BN_SCALE) + bn_b  ==  dot*scale + bias
__global__ void make_scale_bias_kernel(const float* __restrict__ g,
                                       const float* __restrict__ bn_b,
                                       const float* __restrict__ lin_b,
                                       int n, int np,
                                       float* __restrict__ scale,
                                       float* __restrict__ bias) {
    int i = blockIdx.x * blockDim.x + threadIdx.x;
    if (i >= np) return;
    if (i < n) {
        float s = g[i] * BN_SCALE;
        scale[i] = s;
        bias[i]  = lin_b[i] * s + bn_b[i];
    } else {
        scale[i] = 0.0f;
        bias[i]  = 0.0f;
    }
}

// --------------------------- COO SpMM (atomic) -----------------------------
// One wave per nnz; H is bf16 (row stride KP), S is f32 accumulator.
__global__ void spmm_coo_kernel(const int* __restrict__ rows,
                                const int* __restrict__ cols,
                                const float* __restrict__ vals, int nnz,
                                const unsigned short* __restrict__ H,
                                float* __restrict__ S) {
    int wave = (int)((blockIdx.x * (long long)blockDim.x + threadIdx.x) >> 5);
    int lane = threadIdx.x & 31;
    if (wave >= nnz) return;
    int   r = rows[wave];
    int   c = cols[wave];
    float v = vals[wave];
    const unsigned short* h = H + (size_t)c * KP;
    float*                s = S + (size_t)r * KP;
    #pragma unroll
    for (int i = 0; i < 10; ++i) {
        int d = lane + 32 * i;
        if (d < DIM) atomicAdd(&s[d], v * bf16_bits_to_f32(h[d]));
    }
}

// --------------------------- WMMA GEMM -------------------------------------
// C(bf16)[M x ldc=KP] = epilogue( A(bf16)[M x KP] @ B(bf16)[Np x KP]^T )
// One wave -> 16x32 output strip (2 N-tiles), A fragment reused.
// K = KP = 320 -> 10 k-steps -> 20 v_wmma per wave, fully unrolled.
__global__ void __launch_bounds__(256)
wmma_gemm_kernel(const unsigned short* __restrict__ A,
                 const unsigned short* __restrict__ B,
                 unsigned short* __restrict__ C,
                 int M, int Ntiles,            // Ntiles must be even
                 const float* __restrict__ scale,
                 const float* __restrict__ bias,
                 int relu) {
    int wave = (int)((blockIdx.x * (long long)blockDim.x + threadIdx.x) >> 5);
    int lane = threadIdx.x & 31;
    int npairs = Ntiles >> 1;
    int mtile  = wave / npairs;
    int pair   = wave - mtile * npairs;
    if (mtile >= (M >> 4)) return;            // wave-uniform: EXEC all-ones below

    const int tile_m = mtile << 4;
    const int tile_n = pair << 5;             // 32 columns per wave
    const int half   = lane >> 4;             // K-half selector
    const int l16    = lane & 15;

    const unsigned short* arow  = A + (size_t)(tile_m + l16) * KP;
    const unsigned short* brow0 = B + (size_t)(tile_n + l16) * KP;        // n-tile 0
    const unsigned short* brow1 = B + (size_t)(tile_n + 16 + l16) * KP;   // n-tile 1

    v8f acc0, acc1;
    #pragma unroll
    for (int i = 0; i < 8; ++i) { acc0[i] = 0.0f; acc1[i] = 0.0f; }

    #pragma unroll
    for (int kt = 0; kt < KP; kt += 32) {
        BFrag a, b0, b1;
        // A: elements e -> k = kt + half*16 + e   (two 16B loads)
        a.h[0] = *(const v8bf*)(arow + kt + half * 16);
        a.h[1] = *(const v8bf*)(arow + kt + half * 16 + 8);
        // B: lane holds column n; elements e -> k = kt + half*16 + e (2x16B)
        b0.h[0] = *(const v8bf*)(brow0 + kt + half * 16);
        b0.h[1] = *(const v8bf*)(brow0 + kt + half * 16 + 8);
        b1.h[0] = *(const v8bf*)(brow1 + kt + half * 16);
        b1.h[1] = *(const v8bf*)(brow1 + kt + half * 16 + 8);

        acc0 = __builtin_amdgcn_wmma_f32_16x16x32_bf16(
                   false, a.v, false, b0.v, (short)0, acc0, false, false);
        acc1 = __builtin_amdgcn_wmma_f32_16x16x32_bf16(
                   false, a.v, false, b1.v, (short)0, acc1, false, false);
    }

    // epilogue: lane l16 = n, VGPR r -> m = half*8 + r ; write bf16
    int n0 = tile_n + l16;
    int n1 = tile_n + 16 + l16;
    float sc0 = scale ? scale[n0] : 1.0f;
    float bi0 = bias  ? bias[n0]  : 0.0f;
    float sc1 = scale ? scale[n1] : 1.0f;
    float bi1 = bias  ? bias[n1]  : 0.0f;
    #pragma unroll
    for (int r = 0; r < 8; ++r) {
        int m = tile_m + half * 8 + r;
        float v0 = acc0[r] * sc0 + bi0;
        float v1 = acc1[r] * sc1 + bi1;
        if (relu) { v0 = fmaxf(v0, 0.0f); v1 = fmaxf(v1, 0.0f); }
        C[(size_t)m * KP + n0] = f32_to_bf16_bits(v0);
        C[(size_t)m * KP + n1] = f32_to_bf16_bits(v1);
    }
}

// ---------------- residual + LayerNorm + (+emb) fused ----------------------
// G(bf16) = layernorm(0.3*emb + 0.7*H3)*g + b + emb   (one wave per word row)
// H3 is bf16 (stride KP); emb is the raw f32 input (stride DIM).
__global__ void res_ln_addemb_kernel(const unsigned short* __restrict__ H3,
                                     const float* __restrict__ emb,
                                     const float* __restrict__ ln_g,
                                     const float* __restrict__ ln_b,
                                     unsigned short* __restrict__ G, int nrows) {
    int wave = (int)((blockIdx.x * (long long)blockDim.x + threadIdx.x) >> 5);
    int lane = threadIdx.x & 31;
    if (wave >= nrows) return;
    const unsigned short* h = H3 + (size_t)wave * KP;
    const float*          e = emb + (size_t)wave * DIM;
    float loc[10], em[10];
    float sum = 0.0f, sumsq = 0.0f;
    #pragma unroll
    for (int i = 0; i < 10; ++i) {
        int d = lane + 32 * i;
        float x = 0.0f, ev = 0.0f;
        if (d < DIM) {
            ev = e[d];
            x  = (1.0f - ALPHA) * ev + ALPHA * bf16_bits_to_f32(h[d]);
        }
        loc[i] = x; em[i] = ev; sum += x; sumsq += x * x;
    }
    #pragma unroll
    for (int off = 16; off > 0; off >>= 1) {
        sum   += __shfl_xor(sum,   off, 32);
        sumsq += __shfl_xor(sumsq, off, 32);
    }
    float mu   = sum * (1.0f / DIM);
    float var  = sumsq * (1.0f / DIM) - mu * mu;
    float rstd = rsqrtf(var + 1e-5f);
    #pragma unroll
    for (int i = 0; i < 10; ++i) {
        int d = lane + 32 * i;
        float out = 0.0f;
        if (d < DIM) out = (loc[i] - mu) * rstd * ln_g[d] + ln_b[d] + em[i];
        G[(size_t)wave * KP + d] = f32_to_bf16_bits(out);  // pads -> bf16 zero
    }
}

// ---------------- classifier head [ND x 2] ---------------------------------
__global__ void clf_kernel(const unsigned short* __restrict__ H2,
                           const float* __restrict__ W,   // [2 x 150] f32
                           const float* __restrict__ b,   // [2]
                           float* __restrict__ out) {
    int t = blockIdx.x * blockDim.x + threadIdx.x;
    if (t >= ND * 2) return;
    int doc = t >> 1, cls = t & 1;
    const unsigned short* h = H2 + (size_t)doc * KP;
    const float* w = W + cls * D2;
    float acc = b[cls];
    for (int k = 0; k < D2; ++k) acc += bf16_bits_to_f32(h[k]) * w[k];
    out[t] = acc;
}

// ---------------------------------------------------------------------------

static inline int cdiv(long long a, long long b) { return (int)((a + b - 1) / b); }

extern "C" void kernel_launch(void* const* d_in, const int* in_sizes, int n_in,
                              void* d_out, int out_size, void* d_ws, size_t ws_size,
                              hipStream_t stream) {
    (void)in_sizes; (void)n_in; (void)out_size; (void)ws_size;
    const int*   a_rows = (const int*)  d_in[0];
    const int*   a_cols = (const int*)  d_in[1];
    const float* a_vals = (const float*)d_in[2];
    const int*   x_rows = (const int*)  d_in[3];
    const int*   x_cols = (const int*)  d_in[4];
    const float* x_vals = (const float*)d_in[5];
    const float* emb    = (const float*)d_in[6];
    const float* w1     = (const float*)d_in[7];
    const float* w2     = (const float*)d_in[8];
    const float* w3     = (const float*)d_in[9];
    const float* ln_g   = (const float*)d_in[10];
    const float* ln_b   = (const float*)d_in[11];
    const float* m_w1   = (const float*)d_in[12];
    const float* m_b1   = (const float*)d_in[13];
    const float* bn1_g  = (const float*)d_in[14];
    const float* bn1_b  = (const float*)d_in[15];
    const float* m_w2   = (const float*)d_in[16];
    const float* m_b2   = (const float*)d_in[17];
    const float* bn2_g  = (const float*)d_in[18];
    const float* bn2_b  = (const float*)d_in[19];
    const float* clf_w  = (const float*)d_in[20];
    const float* clf_b  = (const float*)d_in[21];
    float* out = (float*)d_out;

    // ---- workspace layout ----
    const size_t R320 = (size_t)NW * KP;           // elements per word-matrix
    char* base = (char*)d_ws;
    float*          S    = (float*)base;                       // f32 accum
    unsigned short* Sb   = (unsigned short*)(base + R320 * 4);           // bf16
    unsigned short* Eb   = Sb + R320;
    unsigned short* HAb  = Eb + R320;
    unsigned short* HBb  = HAb + R320;
    unsigned short* W1b  = HBb + R320;                         // [320x320]
    unsigned short* W2b  = W1b + (size_t)KP * KP;
    unsigned short* W3b  = W2b + (size_t)KP * KP;
    unsigned short* MW1b = W3b + (size_t)KP * KP;              // [320x320]
    unsigned short* MW2b = MW1b + (size_t)KP * KP;             // [160x320]
    float* sc1 = (float*)(MW2b + (size_t)NP2 * KP);
    float* bi1 = sc1 + KP;
    float* sc2 = bi1 + KP;
    float* bi2 = sc2 + NP2;

    const int BT = 256;

    // ---- stage 0: pad+convert weights & emb to bf16; fold BN scale/bias ----
    pad_cvt_bf16_kernel<<<cdiv(R320, BT), BT, 0, stream>>>(emb, NW, DIM, Eb, NW, KP);
    pad_cvt_bf16_kernel<<<cdiv((size_t)KP * KP, BT), BT, 0, stream>>>(w1,   DIM, DIM, W1b,  KP,  KP);
    pad_cvt_bf16_kernel<<<cdiv((size_t)KP * KP, BT), BT, 0, stream>>>(w2,   DIM, DIM, W2b,  KP,  KP);
    pad_cvt_bf16_kernel<<<cdiv((size_t)KP * KP, BT), BT, 0, stream>>>(w3,   DIM, DIM, W3b,  KP,  KP);
    pad_cvt_bf16_kernel<<<cdiv((size_t)KP * KP, BT), BT, 0, stream>>>(m_w1, DIM, DIM, MW1b, KP,  KP);
    pad_cvt_bf16_kernel<<<cdiv((size_t)NP2 * KP, BT), BT, 0, stream>>>(m_w2, D2, DIM, MW2b, NP2, KP);
    make_scale_bias_kernel<<<cdiv(KP, BT),  BT, 0, stream>>>(bn1_g, bn1_b, m_b1, DIM, KP,  sc1, bi1);
    make_scale_bias_kernel<<<cdiv(NP2, BT), BT, 0, stream>>>(bn2_g, bn2_b, m_b2, D2,  NP2, sc2, bi2);

    // ---- stage 1: word GCN, 3x (SpMM -> cvt -> WMMA GEMM + ReLU) ----
    const int spmmA_blocks = cdiv((long long)NNZA * 32, BT);
    const int gcn_waves    = (NW / 16) * (KP / 32);            // 1875 * 10 pairs
    const int gcn_blocks   = cdiv((long long)gcn_waves * 32, BT);

    const unsigned short* Hsrc = Eb;
    unsigned short* Hdst = HAb;
    const unsigned short* Wmat[3] = { W1b, W2b, W3b };
    for (int layer = 0; layer < 3; ++layer) {
        zero_f32_kernel<<<cdiv(R320, BT), BT, 0, stream>>>(S, (long long)R320);
        spmm_coo_kernel<<<spmmA_blocks, BT, 0, stream>>>(a_rows, a_cols, a_vals,
                                                         NNZA, Hsrc, S);
        cvt_f32_bf16_kernel<<<cdiv(R320 / 8, BT), BT, 0, stream>>>(S, Sb, (long long)R320);
        wmma_gemm_kernel<<<gcn_blocks, BT, 0, stream>>>(Sb, Wmat[layer], Hdst,
                                                        NW, KP / 16,
                                                        nullptr, nullptr, /*relu=*/1);
        Hsrc = Hdst;
        Hdst = (Hdst == HAb) ? HBb : HAb;
    }
    // after loop: h3 lives in HAb (Eb->HAb, HAb->HBb, HBb->HAb)

    // ---- stage 2: residual + LayerNorm fused (+emb): G -> HBb ----
    res_ln_addemb_kernel<<<cdiv((long long)NW * 32, BT), BT, 0, stream>>>(
        HAb, emb, ln_g, ln_b, HBb, NW);

    // ---- stage 3: doc aggregation: Dacc = X @ G ----
    const size_t DROWS = (size_t)ND * KP;
    zero_f32_kernel<<<cdiv(DROWS, BT), BT, 0, stream>>>(S, (long long)DROWS);
    spmm_coo_kernel<<<cdiv((long long)NNZX * 32, BT), BT, 0, stream>>>(
        x_rows, x_cols, x_vals, NNZX, HBb, S);
    cvt_f32_bf16_kernel<<<cdiv(DROWS / 8, BT), BT, 0, stream>>>(S, Sb, (long long)DROWS);

    // ---- stage 4: MLP head (WMMA + folded BN + ReLU) ----
    const int mlp1_waves = (ND / 16) * (KP / 32);      // 512 * 10 pairs
    wmma_gemm_kernel<<<cdiv((long long)mlp1_waves * 32, BT), BT, 0, stream>>>(
        Sb, MW1b, HAb, ND, KP / 16, sc1, bi1, /*relu=*/1);
    const int mlp2_waves = (ND / 16) * (NP2 / 32);     // 512 * 5 pairs
    wmma_gemm_kernel<<<cdiv((long long)mlp2_waves * 32, BT), BT, 0, stream>>>(
        HAb, MW2b, HBb, ND, NP2 / 16, sc2, bi2, /*relu=*/1);

    // ---- stage 5: classifier ----
    clf_kernel<<<cdiv(ND * 2, BT), BT, 0, stream>>>(HBb, clf_w, clf_b, out);
}
